// RNN_73538430042419
// MI455X (gfx1250) — matrix operand order repaired and checked by reading, not compiled, not run on hardware
//
#include <hip/hip_runtime.h>
#include <hip/hip_bf16.h>
#include <stdint.h>

// ---------------------------------------------------------------------------
// LSTM (B=8192, T=128, F=64, H=256) + MLP head, persistent-workgroup design.
//
// Per step: g = [x_t | h] @ Wg^T + bg with Wg (1024 x 320).
//   - One workgroup owns 16 batch rows for all 128 steps (h,c live in LDS).
//   - GEMM tile: M=16 (batch rows), N=1024 (gate cols, 128 per wave),
//     K=320 in 10 chunks of 32, via v_wmma_f32_16x16x32_bf16.
//   - Weights re-packed once into bf16 B-operand layout in d_ws, then served
//     from the 192MB L2 (655KB working set) every step.
//   - K-loop is peeled (inline-0 C operand) + kept rolled to avoid VGPR
//     spills; bias is folded into the elementwise gate phase.
// ---------------------------------------------------------------------------

typedef __attribute__((ext_vector_type(16))) __bf16 v16bf;
typedef __attribute__((ext_vector_type(8)))  float  v8f;

#define B_BATCH 8192
#define T_STEPS 128
#define F_IN    64
#define H_DIM   256
#define FH      320     // F_IN + H_DIM (K dimension)
#define G4      1024    // 4*H gate columns (N dimension)
#define KCHUNKS 10      // FH / 32
#define M_TILE  16
#define WG_THREADS 256
#define NUM_WG  (B_BATCH / M_TILE)   // 512

__device__ __forceinline__ unsigned short f32_to_bf16_bits(float f) {
  uint32_t u = __float_as_uint(f);
  uint32_t r = (u + 0x7FFFu + ((u >> 16) & 1u)) >> 16;   // round-nearest-even
  return (unsigned short)r;
}
__device__ __forceinline__ float bf16_bits_to_f32(unsigned short b) {
  return __uint_as_float(((uint32_t)b) << 16);
}
__device__ __forceinline__ float sigmoidf_(float x) {
  return 1.0f / (1.0f + __expf(-x));
}

// A-operand (16-bit, 16x32) VGPR layout per ISA 7.12.2:
//   lanes 0-15: M rows; halves 0..7 -> K 0..7, halves 8..15 -> K 16..23
//   lanes 16-31: halves 0..7 -> K 8..15, halves 8..15 -> K 24..31
// Slot index (in ushorts) of element (m, k) in the packed zA LDS buffer.
__device__ __forceinline__ int zA_slot(int m, int k) {
  int kc = k >> 5;
  int kk = k & 31;
  int lg = (kk >> 3) & 1;                      // lane group
  int hh = (kk & 7) + ((kk >> 4) << 3);        // half index within lane
  int lane = (lg << 4) + m;
  return ((kc * 32 + lane) << 4) + hh;
}

// ---------------------------------------------------------------------------
// Kernel 1: pack Wg = concat(Wf,Wu,Wc,Wo) (f32, row-major (1024 x 320)) into
// bf16 B-operand fragments:  wp[((tile*10 + kc)*32 + lane)*16 + half]
// B-operand (32x16) layout: lane = N col (0-15 / 16-31 mirror),
// lanes 0-15 hold K 0..7,16..23 in halves 0..15, lanes 16-31 hold K 8..15,24..31.
// ---------------------------------------------------------------------------
__global__ void pack_gate_weights(const float* __restrict__ Wf,
                                  const float* __restrict__ Wu,
                                  const float* __restrict__ Wc,
                                  const float* __restrict__ Wo,
                                  unsigned short* __restrict__ wp) {
  int p = blockIdx.x * blockDim.x + threadIdx.x;
  if (p >= G4 * FH) return;
  int half = p & 15;
  int lane = (p >> 4) & 31;
  int kc   = (p >> 9) % KCHUNKS;
  int tile = p / (KCHUNKS * 512);
  int n = tile * 16 + (lane & 15);             // gate column 0..1023
  int k = kc * 32 + ((lane >> 4) << 3) + (half & 7) + ((half >> 3) << 4); // table K map for B operand
  const float* W = (n < 256) ? Wf : (n < 512) ? Wu : (n < 768) ? Wc : Wo;
  int j = n & 255;
  wp[p] = f32_to_bf16_bits(W[j * FH + k]);
}

// ---------------------------------------------------------------------------
// Kernel 2: persistent LSTM + head. One WG = 16 batch rows, 8 wave32s.
// ---------------------------------------------------------------------------
__global__ void __launch_bounds__(WG_THREADS, 1)
lstm_persistent(const float* __restrict__ x,
                const float* __restrict__ pred_month,
                const float* __restrict__ latlons,
                const float* __restrict__ bfp, const float* __restrict__ bup,
                const float* __restrict__ bcp, const float* __restrict__ bop,
                const float* __restrict__ W1, const float* __restrict__ b1,
                const float* __restrict__ W2, const float* __restrict__ b2,
                const float* __restrict__ Wfin, const float* __restrict__ bfin,
                const unsigned short* __restrict__ wp,
                float* __restrict__ out) {
  __shared__ __attribute__((aligned(32))) unsigned short zA[KCHUNKS * 32 * 16]; // 10 KB
  __shared__ _Float16 gbuf[M_TILE * G4];                                        // 32 KB
  __shared__ float    cL[M_TILE * H_DIM];                                       // 16 KB
  __shared__ float    biasL[G4];                                                //  4 KB

  const int tid  = threadIdx.x;
  const int lane = tid & 31;          // wave32
  const int wave = tid >> 5;          // 8 waves
  const int r0   = blockIdx.x * M_TILE;

  // ---- init: h=0 (zA gate region), c=0, biases staged in LDS ----
  for (int i = tid; i < KCHUNKS * 32 * 16; i += WG_THREADS) zA[i] = 0;
  for (int i = tid; i < M_TILE * H_DIM; i += WG_THREADS) cL[i] = 0.0f;
  for (int i = tid; i < G4; i += WG_THREADS) {
    int gate = i >> 8, j = i & 255;
    const float* bp = (gate == 0) ? bfp : (gate == 1) ? bup : (gate == 2) ? bcp : bop;
    biasL[i] = bp[j];
  }
  __syncthreads();

  const unsigned short* wpWave = wp + (size_t)(wave * 8) * (KCHUNKS * 512);

  for (int t = 0; t < T_STEPS; ++t) {
    // ---- phase A: stage x_t (16x64) into packed A layout ----
    for (int e = tid; e < M_TILE * F_IN; e += WG_THREADS) {
      int m = e >> 6, k = e & 63;
      size_t gidx = (size_t)(r0 + m) * (T_STEPS * F_IN) + (size_t)t * F_IN + k;
      zA[zA_slot(m, k)] = f32_to_bf16_bits(x[gidx]);
      if (t + 1 < T_STEPS)
        __builtin_prefetch(&x[gidx + F_IN], 0, 1);   // global_prefetch_b8 next step
    }
    __syncthreads();

    // ---- phase B: 16x1024x320 GEMM; wave owns 128 gate columns ----
    v8f acc[8];
    // Peeled kc = 0: C operand is literal zero (backend can use inline SRC2=0).
    {
      v16bf afrag = *(const v16bf*)(&zA[lane << 4]);
      const v8f zero = {0.f, 0.f, 0.f, 0.f, 0.f, 0.f, 0.f, 0.f};
      const unsigned short* wpk = wpWave + lane * 16;
      #pragma unroll
      for (int tl = 0; tl < 8; ++tl) {
        v16bf bfrag = *(const v16bf*)(wpk + (size_t)tl * (KCHUNKS * 512));
        acc[tl] = __builtin_amdgcn_wmma_f32_16x16x32_bf16(
            false, afrag, false, bfrag, (short)0, zero, false, false);
      }
    }
    // Remaining K chunks: keep the loop rolled so only one A frag and a few
    // pipelined B frags are live at a time (no VGPR spills).
    #pragma unroll 1
    for (int kc = 1; kc < KCHUNKS; ++kc) {
      v16bf afrag = *(const v16bf*)(&zA[(kc * 32 + lane) << 4]);  // 2x ds_load_b128
      const unsigned short* wpk = wpWave + (size_t)kc * 512 + lane * 16;
      #pragma unroll
      for (int tl = 0; tl < 8; ++tl) {
        v16bf bfrag = *(const v16bf*)(wpk + (size_t)tl * (KCHUNKS * 512));
        acc[tl] = __builtin_amdgcn_wmma_f32_16x16x32_bf16(
            false, afrag, false, bfrag, (short)0, acc[tl], false, false);
      }
    }
    // C layout: VGPR r, lanes 0-15 -> M=r, lanes 16-31 -> M=8+r; N = lane&15.
    {
      int Mrow  = (lane >> 4) << 3;
      int Nbase = wave * 128 + (lane & 15);
      #pragma unroll
      for (int tl = 0; tl < 8; ++tl)
        #pragma unroll
        for (int r = 0; r < 8; ++r)
          gbuf[(Mrow + r) * G4 + Nbase + tl * 16] = (_Float16)acc[tl][r];
    }
    __syncthreads();

    // ---- phase C: bias + gate nonlinearity, update c, write h into zA ----
    for (int e = tid; e < M_TILE * H_DIM; e += WG_THREADS) {
      int m = e >> 8, j = e & 255;
      float gf = (float)gbuf[m * G4 + j]       + biasL[j];
      float gu = (float)gbuf[m * G4 + 256 + j] + biasL[256 + j];
      float gc = (float)gbuf[m * G4 + 512 + j] + biasL[512 + j];
      float go = (float)gbuf[m * G4 + 768 + j] + biasL[768 + j];
      float f = sigmoidf_(gf);
      float u = sigmoidf_(gu);
      float o = sigmoidf_(go);
      float cn = f * cL[e] + u * tanhf(gc);
      cL[e] = cn;
      float h = o * tanhf(cn);
      zA[zA_slot(m, F_IN + j)] = f32_to_bf16_bits(h);
    }
    __syncthreads();
  }

  // ---- head MLP (tiny: ~0.1 MFLOP per row) ----
  // feat1 = relu([h, pred_month, latlons] @ W1^T + b1)  -> reuse cL (16x256 f32)
  float* feat1 = cL;
  for (int e = tid; e < M_TILE * 256; e += WG_THREADS) {
    int m = e >> 8, o = e & 255;
    const float* wrow = W1 + o * 270;
    float s = b1[o];
    for (int i = 0; i < 256; ++i)
      s += bf16_bits_to_f32(zA[zA_slot(m, F_IN + i)]) * wrow[i];
    int r = r0 + m;
    for (int i = 0; i < 12; ++i) s += pred_month[r * 12 + i] * wrow[256 + i];
    s += latlons[r * 2 + 0] * wrow[268];
    s += latlons[r * 2 + 1] * wrow[269];
    feat1[m * 256 + o] = fmaxf(s, 0.0f);
  }
  __syncthreads();

  // feat2 = relu(feat1 @ W2^T + b2) -> reuse gbuf storage as f32 (16x128)
  float* feat2 = (float*)gbuf;
  for (int e = tid; e < M_TILE * 128; e += WG_THREADS) {
    int m = e >> 7, o = e & 127;
    const float* wrow = W2 + o * 256;
    float s = b2[o];
    for (int i = 0; i < 256; ++i) s += feat1[m * 256 + i] * wrow[i];
    feat2[m * 128 + o] = fmaxf(s, 0.0f);
  }
  __syncthreads();

  if (tid < M_TILE) {
    int m = tid;
    float s = bfin[0];
    for (int i = 0; i < 128; ++i) s += feat2[m * 128 + i] * Wfin[i];
    out[r0 + m] = s;
  }
}

// ---------------------------------------------------------------------------
extern "C" void kernel_launch(void* const* d_in, const int* in_sizes, int n_in,
                              void* d_out, int out_size, void* d_ws, size_t ws_size,
                              hipStream_t stream) {
  (void)in_sizes; (void)n_in; (void)out_size; (void)ws_size;
  const float* x    = (const float*)d_in[0];
  const float* pm   = (const float*)d_in[1];
  const float* ll   = (const float*)d_in[2];
  const float* Wf   = (const float*)d_in[3];
  const float* bfp  = (const float*)d_in[4];
  const float* Wu   = (const float*)d_in[5];
  const float* bup  = (const float*)d_in[6];
  const float* Wc   = (const float*)d_in[7];
  const float* bcp  = (const float*)d_in[8];
  const float* Wo   = (const float*)d_in[9];
  const float* bop  = (const float*)d_in[10];
  const float* W1   = (const float*)d_in[11];
  const float* b1   = (const float*)d_in[12];
  const float* W2   = (const float*)d_in[13];
  const float* b2   = (const float*)d_in[14];
  const float* Wfin = (const float*)d_in[15];
  const float* bfin = (const float*)d_in[16];

  unsigned short* wp = (unsigned short*)d_ws;   // 1024*320*2 = 655,360 B

  pack_gate_weights<<<(G4 * FH + 255) / 256, 256, 0, stream>>>(Wf, Wu, Wc, Wo, wp);
  lstm_persistent<<<NUM_WG, WG_THREADS, 0, stream>>>(
      x, pm, ll, bfp, bup, bcp, bop, W1, b1, W2, b2, Wfin, bfin, wp,
      (float*)d_out);
}
